// PairUpdate_5531917877751
// MI455X (gfx1250) — matrix operand order repaired and checked by reading, not compiled, not run on hardware
//
#include <hip/hip_runtime.h>
#include <math.h>

// ---------------------------------------------------------------------------
// AlphaFold-style pair block for MI455X (gfx1250, wave32, WMMA 16x16x32 bf16)
// All GEMMs run as bf16 x bf16 -> f32 WMMA; elementwise math in f32.
// Heavy GEMMs: 128x64 block tile, 32x32 per wave (4 WMMA/fragment set),
// double-buffered LDS fed by CDNA5 async global->LDS copies (ASYNCcnt),
// one barrier per K-step.
// ---------------------------------------------------------------------------

#define LL   256
#define PP   65536      // LL*LL
#define DIMC 128

typedef __attribute__((ext_vector_type(16))) __bf16 v16bf;
typedef __attribute__((ext_vector_type(8)))  __bf16 v8bf;
typedef __attribute__((ext_vector_type(4)))  __bf16 v4bf;
typedef __attribute__((ext_vector_type(8)))  float  v8f;

static __device__ __forceinline__ float  bf2f(__bf16 x) { return (float)x; }
static __device__ __forceinline__ __bf16 f2bf(float x)  { return (__bf16)x; }
static __device__ __forceinline__ float  sigf(float x)  { return 1.0f / (1.0f + __expf(-x)); }

static __device__ __forceinline__ v8f wmma_bf16(v16bf a, v16bf b, v8f c)
{
    return __builtin_amdgcn_wmma_f32_16x16x32_bf16(false, a, false, b,
                                                   (short)0, c, false, false);
}

// CDNA5 async global->LDS copy, 16B per lane.  LDS address = low 32 bits of
// the flat pointer (ISA 10.2: LDS aperture truncates to addr[31:0]).
static __device__ __forceinline__ void async_cp16(const __bf16* g, __bf16* l)
{
    const unsigned laddr = (unsigned)(unsigned long long)l;
    asm volatile("global_load_async_to_lds_b128 %0, %1, off"
                 :: "v"(laddr), "v"(g) : "memory");
}
static __device__ __forceinline__ void async_wait0()
{
    asm volatile("s_wait_asynccnt 0x0" ::: "memory");
}

// ---------------------------------------------------------------------------
// LayerNorm over last dim (128), f32 in -> bf16 out.  One wave per row.
// ---------------------------------------------------------------------------
__global__ __launch_bounds__(256) void ln_rows_kernel(
    const float* __restrict__ src, const float* __restrict__ gw,
    const float* __restrict__ bw, __bf16* __restrict__ dst)
{
    const int wave = threadIdx.x >> 5, lane = threadIdx.x & 31;
    const int row  = blockIdx.x * 8 + wave;
    const float4 x = *(const float4*)(src + (size_t)row * DIMC + lane * 4);
    float s  = x.x + x.y + x.z + x.w;
    float sq = x.x*x.x + x.y*x.y + x.z*x.z + x.w*x.w;
    #pragma unroll
    for (int off = 16; off >= 1; off >>= 1) {
        s  += __shfl_xor(s,  off, 32);
        sq += __shfl_xor(sq, off, 32);
    }
    const float mean = s * (1.0f / DIMC);
    const float var  = sq * (1.0f / DIMC) - mean * mean;
    const float rstd = rsqrtf(var + 1e-5f);
    const float4 gg = *(const float4*)(gw + lane * 4);
    const float4 bb = *(const float4*)(bw + lane * 4);
    v4bf o;
    o[0] = f2bf((x.x - mean) * rstd * gg.x + bb.x);
    o[1] = f2bf((x.y - mean) * rstd * gg.y + bb.y);
    o[2] = f2bf((x.z - mean) * rstd * gg.z + bb.z);
    o[3] = f2bf((x.w - mean) * rstd * gg.w + bb.w);
    *(v4bf*)(dst + (size_t)row * DIMC + lane * 4) = o;
}

// ---------------------------------------------------------------------------
// Weight prep: f32 W[K][N] -> bf16 WT[N][K]
// ---------------------------------------------------------------------------
__global__ void wprep_kernel(const float* __restrict__ src,
                             __bf16* __restrict__ dst, int K, int N)
{
    const int t = blockIdx.x * 256 + threadIdx.x;
    if (t >= K * N) return;
    const int n = t / K, k = t - n * K;
    dst[t] = f2bf(src[k * N + n]);
}

// ---------------------------------------------------------------------------
// WMMA fragment loaders (layouts per CDNA5 ISA 7.12.2, 16-bit A 16x32 / B 32x16)
// LDS row stride 40 bf16 (80B) to dodge bank conflicts.
// ---------------------------------------------------------------------------
static __device__ __forceinline__ v16bf frag_a(const __bf16* As, int mi, int lane)
{
    const int m  = mi * 16 + (lane & 15);
    const int kh = (lane < 16) ? 0 : 8;
    union { v16bf v; v8bf h[2]; } f;
    f.h[0] = *(const v8bf*)(As + m * 40 + kh);
    f.h[1] = *(const v8bf*)(As + m * 40 + 16 + kh);
    return f.v;
}
static __device__ __forceinline__ v16bf frag_b(const __bf16* Bs, int ni, int lane)
{
    const int n  = ni * 16 + (lane & 15);
    const int kb = (lane < 16) ? 0 : 8;
    union { v16bf v; v8bf h[2]; } f;
    f.h[0] = *(const v8bf*)(Bs + n * 40 + kb);
    f.h[1] = *(const v8bf*)(Bs + n * 40 + 16 + kb);
    return f.v;
}

enum { EPI_F32 = 0, EPI_BF16 = 1, EPI_SIG = 2, EPI_GATE = 3, EPI_FINAL = 4 };

// ---------------------------------------------------------------------------
// Heavy GEMM: C[M,N] = A[M,K] * BT[N,K]^T, block tile 128x64, wave tile 32x32,
// double-buffered LDS fed by async copies.  grid = (N/64, M/128, channels).
// ---------------------------------------------------------------------------
template <int EPI>
__global__ __launch_bounds__(256) void gemm_big_kernel(
    const __bf16* __restrict__ A, const __bf16* __restrict__ BT,
    int K, int lda, int ldb, int ldc,
    float* __restrict__ Cf, __bf16* __restrict__ Cb,
    const float* __restrict__ gate, const float* __restrict__ pre,
    float* __restrict__ post, const float* __restrict__ bias,
    const float* __restrict__ mask,
    long long sA, long long sB, long long sC)
{
    __shared__ __align__(16) __bf16 As[2][128 * 40];
    __shared__ __align__(16) __bf16 Bs[2][64 * 40];
    const int t = threadIdx.x, wave = t >> 5, lane = t & 31;
    const int mi = wave >> 1, ni = wave & 1;          // mi 0..3, ni 0..1
    const int m0 = blockIdx.y * 128, n0 = blockIdx.x * 64;
    const long long c = blockIdx.z;
    A  += c * sA;
    BT += c * sB;
    const int arow = t >> 1, akq = (t & 1) * 16;      // 128 x 32 tile, 16 el/thr
    const int brow = t >> 2, bkq = (t & 3) * 8;       // 64 x 32 tile,  8 el/thr
    const __bf16* Ap = A  + (size_t)(m0 + arow) * lda + akq;
    const __bf16* Bp = BT + (size_t)(n0 + brow) * ldb + bkq;

    // preload tile 0 via async copies
    async_cp16(Ap,     &As[0][arow * 40 + akq]);
    async_cp16(Ap + 8, &As[0][arow * 40 + akq + 8]);
    async_cp16(Bp,     &Bs[0][brow * 40 + bkq]);
    async_wait0();
    __syncthreads();

    v8f acc[2][2] = {};
    const int nsteps = K >> 5;
    for (int s = 0; s < nsteps; ++s) {
        const int cur = s & 1, nxt = cur ^ 1;
        const bool more = (s + 1) < nsteps;
        if (more) {
            const int kb = (s + 1) * 32;
            async_cp16(Ap + kb,     &As[nxt][arow * 40 + akq]);
            async_cp16(Ap + kb + 8, &As[nxt][arow * 40 + akq + 8]);
            async_cp16(Bp + kb,     &Bs[nxt][brow * 40 + bkq]);
        }
        const v16bf af0 = frag_a(&As[cur][0], 2 * mi,     lane);
        const v16bf af1 = frag_a(&As[cur][0], 2 * mi + 1, lane);
        const v16bf bf0 = frag_b(&Bs[cur][0], 2 * ni,     lane);
        const v16bf bf1 = frag_b(&Bs[cur][0], 2 * ni + 1, lane);
        acc[0][0] = wmma_bf16(af0, bf0, acc[0][0]);
        acc[0][1] = wmma_bf16(af0, bf1, acc[0][1]);
        acc[1][0] = wmma_bf16(af1, bf0, acc[1][0]);
        acc[1][1] = wmma_bf16(af1, bf1, acc[1][1]);
        if (more) async_wait0();
        __syncthreads();
    }

    #pragma unroll
    for (int am = 0; am < 2; ++am)
    #pragma unroll
    for (int bn = 0; bn < 2; ++bn)
    #pragma unroll
    for (int r = 0; r < 8; ++r) {
        const int m = m0 + mi * 32 + am * 16 + r + ((lane < 16) ? 0 : 8);
        const int n = n0 + ni * 32 + bn * 16 + (lane & 15);
        const float v = acc[am][bn][r];
        const size_t idx = (size_t)(c * sC) + (size_t)m * ldc + n;
        if (EPI == EPI_F32)   Cf[idx] = v;
        if (EPI == EPI_BF16)  Cb[idx] = f2bf(v);
        if (EPI == EPI_SIG)   Cf[idx] = sigf(v);
        if (EPI == EPI_GATE)  post[idx] = pre[idx] + v * gate[idx] * mask[m];
        if (EPI == EPI_FINAL) post[idx] = (v + bias[n]) * mask[m];
    }
}

// ---------------------------------------------------------------------------
// Heavy dual GEMM (two B matrices, fused gating): same 128x64 / 32x32 tiling,
// async double-buffered, 8 WMMA per wave per K-step.
// SILU=0: out = sigmoid(acc1)*acc2   SILU=1: out = silu(acc1+b1)*(acc2+b2)
// ---------------------------------------------------------------------------
template <int SILU>
__global__ __launch_bounds__(256) void dual_gemm_big_kernel(
    const __bf16* __restrict__ A, const __bf16* __restrict__ BT1,
    const __bf16* __restrict__ BT2, int K, int lda, int ldb, int ldc,
    __bf16* __restrict__ Cb,
    const float* __restrict__ bias1, const float* __restrict__ bias2)
{
    __shared__ __align__(16) __bf16 As[2][128 * 40];
    __shared__ __align__(16) __bf16 Bs1[2][64 * 40];
    __shared__ __align__(16) __bf16 Bs2[2][64 * 40];
    const int t = threadIdx.x, wave = t >> 5, lane = t & 31;
    const int mi = wave >> 1, ni = wave & 1;
    const int m0 = blockIdx.y * 128, n0 = blockIdx.x * 64;
    const int arow = t >> 1, akq = (t & 1) * 16;
    const int brow = t >> 2, bkq = (t & 3) * 8;
    const __bf16* Ap  = A   + (size_t)(m0 + arow) * lda + akq;
    const __bf16* B1p = BT1 + (size_t)(n0 + brow) * ldb + bkq;
    const __bf16* B2p = BT2 + (size_t)(n0 + brow) * ldb + bkq;

    async_cp16(Ap,     &As[0][arow * 40 + akq]);
    async_cp16(Ap + 8, &As[0][arow * 40 + akq + 8]);
    async_cp16(B1p,    &Bs1[0][brow * 40 + bkq]);
    async_cp16(B2p,    &Bs2[0][brow * 40 + bkq]);
    async_wait0();
    __syncthreads();

    v8f acc1[2][2] = {}, acc2[2][2] = {};
    const int nsteps = K >> 5;
    for (int s = 0; s < nsteps; ++s) {
        const int cur = s & 1, nxt = cur ^ 1;
        const bool more = (s + 1) < nsteps;
        if (more) {
            const int kb = (s + 1) * 32;
            async_cp16(Ap + kb,     &As[nxt][arow * 40 + akq]);
            async_cp16(Ap + kb + 8, &As[nxt][arow * 40 + akq + 8]);
            async_cp16(B1p + kb,    &Bs1[nxt][brow * 40 + bkq]);
            async_cp16(B2p + kb,    &Bs2[nxt][brow * 40 + bkq]);
        }
        const v16bf af0 = frag_a(&As[cur][0], 2 * mi,     lane);
        const v16bf af1 = frag_a(&As[cur][0], 2 * mi + 1, lane);
        const v16bf b10 = frag_b(&Bs1[cur][0], 2 * ni,     lane);
        const v16bf b11 = frag_b(&Bs1[cur][0], 2 * ni + 1, lane);
        const v16bf b20 = frag_b(&Bs2[cur][0], 2 * ni,     lane);
        const v16bf b21 = frag_b(&Bs2[cur][0], 2 * ni + 1, lane);
        acc1[0][0] = wmma_bf16(af0, b10, acc1[0][0]);
        acc1[0][1] = wmma_bf16(af0, b11, acc1[0][1]);
        acc1[1][0] = wmma_bf16(af1, b10, acc1[1][0]);
        acc1[1][1] = wmma_bf16(af1, b11, acc1[1][1]);
        acc2[0][0] = wmma_bf16(af0, b20, acc2[0][0]);
        acc2[0][1] = wmma_bf16(af0, b21, acc2[0][1]);
        acc2[1][0] = wmma_bf16(af1, b20, acc2[1][0]);
        acc2[1][1] = wmma_bf16(af1, b21, acc2[1][1]);
        if (more) async_wait0();
        __syncthreads();
    }

    #pragma unroll
    for (int am = 0; am < 2; ++am)
    #pragma unroll
    for (int bn = 0; bn < 2; ++bn)
    #pragma unroll
    for (int r = 0; r < 8; ++r) {
        const int m = m0 + mi * 32 + am * 16 + r + ((lane < 16) ? 0 : 8);
        const int n = n0 + ni * 32 + bn * 16 + (lane & 15);
        float a = acc1[am][bn][r], b = acc2[am][bn][r], o;
        if (SILU) { a += bias1[n]; b += bias2[n]; o = a * sigf(a) * b; }
        else      { o = sigf(a) * b; }
        Cb[(size_t)m * ldc + n] = f2bf(o);
    }
}

// ---------------------------------------------------------------------------
// Small GEMM (64x32 tile) for odd N (qkv projection, N=96).
// ---------------------------------------------------------------------------
template <int EPI>
__global__ __launch_bounds__(256) void gemm_kernel(
    const __bf16* __restrict__ A, const __bf16* __restrict__ BT,
    int K, int lda, int ldb, int ldc,
    float* __restrict__ Cf, __bf16* __restrict__ Cb)
{
    __shared__ __align__(16) __bf16 As[64 * 40];
    __shared__ __align__(16) __bf16 Bs[32 * 40];
    const int t = threadIdx.x, wave = t >> 5, lane = t & 31;
    const int mi = wave >> 1, ni = wave & 1;
    const int m0 = blockIdx.y * 64, n0 = blockIdx.x * 32;
    const int arow = t >> 2, akq = (t & 3) * 8;
    const int brow = t >> 3, bkq = (t & 7) * 4;
    v8f acc = {};
    for (int kb = 0; kb < K; kb += 32) {
        __syncthreads();
        *(v8bf*)(As + arow * 40 + akq) =
            *(const v8bf*)(A + (size_t)(m0 + arow) * lda + kb + akq);
        *(v4bf*)(Bs + brow * 40 + bkq) =
            *(const v4bf*)(BT + (size_t)(n0 + brow) * ldb + kb + bkq);
        if (kb + 32 < K)
            __builtin_prefetch(A + (size_t)(m0 + arow) * lda + kb + 32 + akq, 0, 0);
        __syncthreads();
        acc = wmma_bf16(frag_a(As, mi, lane), frag_b(Bs, ni, lane), acc);
    }
    #pragma unroll
    for (int r = 0; r < 8; ++r) {
        const int m = m0 + mi * 16 + r + ((lane < 16) ? 0 : 8);
        const int n = n0 + ni * 16 + (lane & 15);
        const float v = acc[r];
        const size_t idx = (size_t)m * ldc + n;
        if (EPI == EPI_F32)  Cf[idx] = v;
        if (EPI == EPI_BF16) Cb[idx] = f2bf(v);
        if (EPI == EPI_SIG)  Cf[idx] = sigf(v);
    }
}

// ---------------------------------------------------------------------------
// channels->planes repack for triangle einsum:
//   outgoing: a_t[c][i][k] = ab[i*L+k][c] * mask[i*L+k]
//   incoming: a_t[c][i][k] = ab[k*L+i][c] * mask[k*L+i]
// (same for b_t from channel 128+c).  grid=(L/32, L, 128/32)
// ---------------------------------------------------------------------------
__global__ __launch_bounds__(256) void c2p_kernel(
    const __bf16* __restrict__ ab, const float* __restrict__ mask,
    __bf16* __restrict__ a_t, __bf16* __restrict__ b_t, int incoming)
{
    __shared__ float tile[32][33];
    const int k0 = blockIdx.x * 32, i = blockIdx.y, c0 = blockIdx.z * 32;
    const int t = threadIdx.x;
    const int kk = t >> 3, cq = (t & 7) * 4;
    const int cc = t >> 3, kq = (t & 7) * 4;
    for (int half = 0; half < 2; ++half) {
        __syncthreads();
        const int kidx = k0 + kk;
        const size_t p = incoming ? ((size_t)kidx * LL + i) : ((size_t)i * LL + kidx);
        const float mv = mask[p];
        const v4bf s4 = *(const v4bf*)(ab + p * 256 + half * 128 + c0 + cq);
        tile[kk][cq + 0] = bf2f(s4[0]) * mv;
        tile[kk][cq + 1] = bf2f(s4[1]) * mv;
        tile[kk][cq + 2] = bf2f(s4[2]) * mv;
        tile[kk][cq + 3] = bf2f(s4[3]) * mv;
        __syncthreads();
        __bf16* dst = half ? b_t : a_t;
        v4bf o;
        o[0] = f2bf(tile[kq + 0][cc]);
        o[1] = f2bf(tile[kq + 1][cc]);
        o[2] = f2bf(tile[kq + 2][cc]);
        o[3] = f2bf(tile[kq + 3][cc]);
        *(v4bf*)(dst + (size_t)(c0 + cc) * PP + (size_t)i * LL + k0 + kq) = o;
    }
}

// ---------------------------------------------------------------------------
// planes->channels transpose with mask:  dst[p][c] = src[c][p] * mask[p]
// grid=(PP/32, 128/32)
// ---------------------------------------------------------------------------
__global__ __launch_bounds__(256) void tpose_pc_kernel(
    const float* __restrict__ src, const float* __restrict__ mask,
    float* __restrict__ dst)
{
    __shared__ float tile[32][33];
    const int p0 = blockIdx.x * 32, c0 = blockIdx.y * 32;
    const int t = threadIdx.x, pl = t & 31, cl = t >> 5;
    #pragma unroll
    for (int it = 0; it < 4; ++it) {
        const int cc = cl + it * 8;
        tile[cc][pl] = src[(size_t)(c0 + cc) * PP + p0 + pl];
    }
    __syncthreads();
    const int p = p0 + (t >> 3), cq = (t & 7) * 4;
    const float mv = mask[p];
    float4 o;
    o.x = tile[cq + 0][t >> 3] * mv;
    o.y = tile[cq + 1][t >> 3] * mv;
    o.z = tile[cq + 2][t >> 3] * mv;
    o.w = tile[cq + 3][t >> 3] * mv;
    *(float4*)(dst + (size_t)p * DIMC + c0 + cq) = o;
}

// ---------------------------------------------------------------------------
// Pair-bias projection: biasmat[p][h] = dot(zln[p,:], W_b[:,h]).  Wave per p.
// ---------------------------------------------------------------------------
__global__ __launch_bounds__(256) void biasproj_kernel(
    const __bf16* __restrict__ zln, const float* __restrict__ Wb,
    float* __restrict__ biasmat)
{
    const int wave = threadIdx.x >> 5, lane = threadIdx.x & 31;
    const int p = blockIdx.x * 8 + wave;
    const v4bf z4 = *(const v4bf*)(zln + (size_t)p * DIMC + lane * 4);
    float acc[4] = {0.f, 0.f, 0.f, 0.f};
    #pragma unroll
    for (int e = 0; e < 4; ++e) {
        const float zv = bf2f(z4[e]);
        const float4 w = *(const float4*)(Wb + (lane * 4 + e) * 4);
        acc[0] += zv * w.x; acc[1] += zv * w.y;
        acc[2] += zv * w.z; acc[3] += zv * w.w;
    }
    #pragma unroll
    for (int off = 16; off >= 1; off >>= 1)
        #pragma unroll
        for (int h = 0; h < 4; ++h) acc[h] += __shfl_xor(acc[h], off, 32);
    if (lane == 0) {
        float4 o = make_float4(acc[0], acc[1], acc[2], acc[3]);
        *(float4*)(biasmat + (size_t)p * 4) = o;
    }
}

// ---------------------------------------------------------------------------
// Triangle attention, starting node.  grid=(H, L); thread = query col j in row i.
// ---------------------------------------------------------------------------
__global__ __launch_bounds__(256) void attn_start_kernel(
    const __bf16* __restrict__ qkv, const float* __restrict__ biasmat,
    const float* __restrict__ mask, __bf16* __restrict__ ao)
{
    __shared__ __align__(16) __bf16 Ks[32 * 8];
    __shared__ __align__(16) __bf16 Vs[32 * 8];
    __shared__ float Ms[32];
    const int h = blockIdx.x, i = blockIdx.y, j = threadIdx.x;
    const float scale = 0.35355339059327373f;  // 1/sqrt(8)
    float q[8];
    {
        const v8bf qv = *(const v8bf*)(qkv + ((size_t)(i * LL + j) * 96 + h * 8));
        #pragma unroll
        for (int d = 0; d < 8; ++d) q[d] = bf2f(qv[d]) * scale;
    }
    const float mq = mask[i * LL + j];
    float mmax = -3.0e38f, lsum = 0.f, o[8] = {0.f};
    for (int kc = 0; kc < LL; kc += 32) {
        __syncthreads();
        {
            const int kk = threadIdx.x >> 3, d = threadIdx.x & 7;
            const size_t pk = (size_t)(i * LL + kc + kk) * 96;
            Ks[kk * 8 + d] = qkv[pk + 32 + h * 8 + d];
            Vs[kk * 8 + d] = qkv[pk + 64 + h * 8 + d];
            if (threadIdx.x < 32) Ms[threadIdx.x] = mask[i * LL + kc + threadIdx.x];
        }
        __syncthreads();
        for (int kk = 0; kk < 32; ++kk) {
            const v8bf kv = *(const v8bf*)(Ks + kk * 8);
            float s = biasmat[((size_t)j * LL + kc + kk) * 4 + h];
            #pragma unroll
            for (int d = 0; d < 8; ++d) s += q[d] * bf2f(kv[d]);
            if (mq * Ms[kk] <= 0.f) s = -1.0e9f;
            const float nm = fmaxf(mmax, s);
            const float corr = __expf(mmax - nm);
            const float pv = __expf(s - nm);
            lsum = lsum * corr + pv;
            const v8bf vv = *(const v8bf*)(Vs + kk * 8);
            #pragma unroll
            for (int d = 0; d < 8; ++d) o[d] = o[d] * corr + pv * bf2f(vv[d]);
            mmax = nm;
        }
    }
    const float inv = 1.0f / lsum;
    v8bf ov;
    #pragma unroll
    for (int d = 0; d < 8; ++d) ov[d] = f2bf(o[d] * inv);
    *(v8bf*)(ao + ((size_t)(i * LL + j) * 32 + h * 8)) = ov;
}

// ---------------------------------------------------------------------------
// Triangle attention, ending node.  grid=(H, L); thread = x.  For output (i,x):
// scores over y: scale*q(i,y).k(x,y) + bias[y*L+i][h]; out = sum_y P * v(y,x).
// ---------------------------------------------------------------------------
__global__ __launch_bounds__(256) void attn_end_kernel(
    const __bf16* __restrict__ qkv, const float* __restrict__ biasmat,
    const float* __restrict__ mask, __bf16* __restrict__ ao)
{
    __shared__ __align__(16) __bf16 Qs[32 * 8];
    __shared__ float Bsh[32];
    const int h = blockIdx.x, i = blockIdx.y, x = threadIdx.x;
    const float scale = 0.35355339059327373f;
    const float mix = mask[i * LL + x];
    float mmax = -3.0e38f, lsum = 0.f, o[8] = {0.f};
    for (int yc = 0; yc < LL; yc += 32) {
        __syncthreads();
        {
            const int yy = threadIdx.x >> 3, d = threadIdx.x & 7;
            Qs[yy * 8 + d] = qkv[(size_t)(i * LL + yc + yy) * 96 + h * 8 + d];
            if (threadIdx.x < 32)
                Bsh[threadIdx.x] = biasmat[((size_t)(yc + threadIdx.x) * LL + i) * 4 + h];
        }
        __syncthreads();
        for (int yy = 0; yy < 32; ++yy) {
            const int y = yc + yy;
            const v8bf kv = *(const v8bf*)(qkv + ((size_t)(x * LL + y) * 96 + 32 + h * 8));
            const v8bf qv = *(const v8bf*)(Qs + yy * 8);
            float s = Bsh[yy];
            #pragma unroll
            for (int d = 0; d < 8; ++d) s += scale * bf2f(qv[d]) * bf2f(kv[d]);
            if (mix * mask[(size_t)y * LL + x] <= 0.f) s = -1.0e9f;
            const float nm = fmaxf(mmax, s);
            const float corr = __expf(mmax - nm);
            const float pv = __expf(s - nm);
            lsum = lsum * corr + pv;
            const v8bf vv = *(const v8bf*)(qkv + ((size_t)(y * LL + x) * 96 + 64 + h * 8));
            #pragma unroll
            for (int d = 0; d < 8; ++d) o[d] = o[d] * corr + pv * bf2f(vv[d]);
            mmax = nm;
        }
    }
    const float inv = 1.0f / lsum;
    v8bf ov;
    #pragma unroll
    for (int d = 0; d < 8; ++d) ov[d] = f2bf(o[d] * inv);
    *(v8bf*)(ao + ((size_t)(i * LL + x) * 32 + h * 8)) = ov;
}

// ---------------------------------------------------------------------------
// Host-side orchestration
// ---------------------------------------------------------------------------
extern "C" void kernel_launch(void* const* d_in, const int* in_sizes, int n_in,
                              void* d_out, int out_size, void* d_ws, size_t ws_size,
                              hipStream_t stream)
{
    (void)in_sizes; (void)n_in; (void)out_size; (void)ws_size;
    // input layout (setup_inputs dict order, params flattened in insertion order)
    // 0 pair, 1 pair_mask,
    // tmo: 2 ng 3 nb 4 W_ab 5 W_abg 6 W_g 7 W_out 8 n1g 9 n1b
    // tmi: 10..17, tas: 18 ng 19 nb 20 W_qkv 21 W_b 22 W_g 23 W_out
    // tae: 24..29, trans: 30 ng 31 nb 32 W_ab 33 b_ab 34 W_out 35 b_out
    const float* mask = (const float*)d_in[1];

    char* ws = (char*)d_ws;
    float*  pair_c = (float*)(ws);                                // 33.6 MB
    __bf16* zln    = (__bf16*)(ws + 33554432);                    // 16.8 MB
    float*  gbuf   = (float*)(ws + 50331648);                     // 33.6 MB
    __bf16* wt     = (__bf16*)(ws + 83886080);                    // ~0.9 MB
    char*   s1     = ws + 84934656;                               // 33.6 MB
    char*   s2     = s1 + 33554432;                               // 33.6 MB
    char*   s3     = s2 + 33554432;                               // 33.6 MB

    // weight-transpose arena offsets (bf16 elements)
    __bf16* wt_tmo_abg = wt;            // 256x128
    __bf16* wt_tmo_ab  = wt + 32768;
    __bf16* wt_tmo_g   = wt + 65536;    // 128x128
    __bf16* wt_tmo_out = wt + 81920;
    __bf16* wt_tmi_abg = wt + 98304;
    __bf16* wt_tmi_ab  = wt + 131072;
    __bf16* wt_tmi_g   = wt + 163840;
    __bf16* wt_tmi_out = wt + 180224;
    __bf16* wt_tas_qkv = wt + 196608;   // 96x128
    __bf16* wt_tas_g   = wt + 208896;
    __bf16* wt_tas_out = wt + 225280;   // 128x32
    __bf16* wt_tae_qkv = wt + 229376;
    __bf16* wt_tae_g   = wt + 241664;
    __bf16* wt_tae_out = wt + 258048;
    __bf16* wt_tr_ab   = wt + 262144;   // 1024x128
    __bf16* wt_tr_out  = wt + 393216;   // 128x512

    hipMemcpyAsync(pair_c, d_in[0], (size_t)PP * DIMC * 4,
                   hipMemcpyDeviceToDevice, stream);

    auto WP = [&](int idx, __bf16* dst, int K, int N) {
        wprep_kernel<<<(K * N) / 256, 256, 0, stream>>>((const float*)d_in[idx], dst, K, N);
    };
    WP(5, wt_tmo_abg, 128, 256);  WP(4, wt_tmo_ab, 128, 256);
    WP(6, wt_tmo_g, 128, 128);    WP(7, wt_tmo_out, 128, 128);
    WP(13, wt_tmi_abg, 128, 256); WP(12, wt_tmi_ab, 128, 256);
    WP(14, wt_tmi_g, 128, 128);   WP(15, wt_tmi_out, 128, 128);
    WP(20, wt_tas_qkv, 128, 96);  WP(22, wt_tas_g, 128, 128);  WP(23, wt_tas_out, 32, 128);
    WP(26, wt_tae_qkv, 128, 96);  WP(28, wt_tae_g, 128, 128);  WP(29, wt_tae_out, 32, 128);
    WP(32, wt_tr_ab, 128, 1024);  WP(34, wt_tr_out, 512, 128);

    const dim3 BLK(256);

    // ---------------- triangle multiplication (outgoing / incoming) -------
    auto tri_mul = [&](int base, __bf16* wabg, __bf16* wab, __bf16* wg,
                       __bf16* wout, int incoming) {
        __bf16* ab   = (__bf16*)s1;                 // P x 256
        __bf16* a_t  = (__bf16*)s2;                 // 128 x 256 x 256
        __bf16* b_t  = (__bf16*)(s2 + 16777216);
        float*  outt = (float*)s3;                  // 128 x 65536
        float*  tmpf = (float*)s1;                  // P x 128 (reuses ab arena)

        ln_rows_kernel<<<PP / 8, BLK, 0, stream>>>(pair_c,
            (const float*)d_in[base + 0], (const float*)d_in[base + 1], zln);
        dual_gemm_big_kernel<0><<<dim3(4, PP / 128, 1), BLK, 0, stream>>>(
            zln, wabg, wab, 128, 128, 128, 256, ab, nullptr, nullptr);
        gemm_big_kernel<EPI_SIG><<<dim3(2, PP / 128, 1), BLK, 0, stream>>>(
            zln, wg, 128, 128, 128, 128, gbuf, nullptr, nullptr, nullptr,
            nullptr, nullptr, nullptr, 0, 0, 0);
        c2p_kernel<<<dim3(8, LL, 4), BLK, 0, stream>>>(ab, mask, a_t, b_t, incoming);
        gemm_big_kernel<EPI_F32><<<dim3(4, 2, 128), BLK, 0, stream>>>(
            a_t, b_t, 256, 256, 256, 256, outt, nullptr, nullptr, nullptr,
            nullptr, nullptr, nullptr, PP, PP, PP);
        tpose_pc_kernel<<<dim3(PP / 32, 4), BLK, 0, stream>>>(outt, mask, tmpf);
        ln_rows_kernel<<<PP / 8, BLK, 0, stream>>>(tmpf,
            (const float*)d_in[base + 6], (const float*)d_in[base + 7], zln);
        gemm_big_kernel<EPI_GATE><<<dim3(2, PP / 128, 1), BLK, 0, stream>>>(
            zln, wout, 128, 128, 128, 128, nullptr, nullptr, gbuf, pair_c,
            pair_c, nullptr, mask, 0, 0, 0);
    };
    tri_mul(2, wt_tmo_abg, wt_tmo_ab, wt_tmo_g, wt_tmo_out, 0);
    tri_mul(10, wt_tmi_abg, wt_tmi_ab, wt_tmi_g, wt_tmi_out, 1);

    // ---------------- triangle attention (starting / ending) --------------
    auto tri_attn = [&](int base, __bf16* wqkv, __bf16* wg, __bf16* wout, int ending) {
        __bf16* qkv     = (__bf16*)s1;                   // P x 96
        float*  biasmat = (float*)(s1 + 12582912);       // P x 4
        __bf16* ao      = (__bf16*)(s1 + 13631488);      // P x 32

        ln_rows_kernel<<<PP / 8, BLK, 0, stream>>>(pair_c,
            (const float*)d_in[base + 0], (const float*)d_in[base + 1], zln);
        gemm_kernel<EPI_BF16><<<dim3(3, PP / 64, 1), BLK, 0, stream>>>(
            zln, wqkv, 128, 128, 128, 96, nullptr, qkv);
        biasproj_kernel<<<PP / 8, BLK, 0, stream>>>(zln,
            (const float*)d_in[base + 3], biasmat);
        gemm_big_kernel<EPI_SIG><<<dim3(2, PP / 128, 1), BLK, 0, stream>>>(
            zln, wg, 128, 128, 128, 128, gbuf, nullptr, nullptr, nullptr,
            nullptr, nullptr, nullptr, 0, 0, 0);
        if (!ending)
            attn_start_kernel<<<dim3(4, LL), BLK, 0, stream>>>(qkv, biasmat, mask, ao);
        else
            attn_end_kernel<<<dim3(4, LL), BLK, 0, stream>>>(qkv, biasmat, mask, ao);
        gemm_big_kernel<EPI_GATE><<<dim3(2, PP / 128, 1), BLK, 0, stream>>>(
            ao, wout, 32, 32, 32, 128, nullptr, nullptr, gbuf, pair_c,
            pair_c, nullptr, mask, 0, 0, 0);
    };
    tri_attn(18, wt_tas_qkv, wt_tas_g, wt_tas_out, 0);
    tri_attn(24, wt_tae_qkv, wt_tae_g, wt_tae_out, 1);

    // ---------------- transition ------------------------------------------
    {
        __bf16* hbuf = (__bf16*)s2;  // P x 512 bf16 spans s2+s3 (64 MB)
        const float* b_ab  = (const float*)d_in[33];
        const float* b_out = (const float*)d_in[35];
        ln_rows_kernel<<<PP / 8, BLK, 0, stream>>>(pair_c,
            (const float*)d_in[30], (const float*)d_in[31], zln);
        dual_gemm_big_kernel<1><<<dim3(8, PP / 128, 1), BLK, 0, stream>>>(
            zln, wt_tr_ab, wt_tr_ab + 512 * 128, 128, 128, 128, 512, hbuf,
            b_ab, b_ab + 512);
        gemm_big_kernel<EPI_FINAL><<<dim3(2, PP / 128, 1), BLK, 0, stream>>>(
            hbuf, wt_tr_out, 512, 512, 512, 128, nullptr, nullptr, nullptr,
            nullptr, (float*)d_out, b_out, mask, 0, 0, 0);
    }
}
